// QDGCNN_semseg_s3dis_61890478735430
// MI455X (gfx1250) — matrix operand order repaired and checked by reading, not compiled, not run on hardware
//
#include <hip/hip_runtime.h>
#include <hip/hip_bf16.h>
#include <stdint.h>

// ---------------------------------------------------------------------------
// CDNA5 (gfx1250) implementation of QDGCNN semseg forward pass.
// GEMMs: V_WMMA_F32_16X16X32_F16, 4 column-tiles register-blocked per wave,
// activation tiles staged to LDS by the Tensor Data Mover, b128 fragment loads.
// ---------------------------------------------------------------------------

typedef __attribute__((ext_vector_type(16))) _Float16 v16h;
typedef __attribute__((ext_vector_type(8)))  float    v8f;
typedef __attribute__((ext_vector_type(4)))  uint32_t u32x4;
typedef __attribute__((ext_vector_type(8)))  uint32_t u32x8;

union Frag { v16h v; uint32_t u[8]; uint4 q[2]; _Float16 h[16]; };

#define NPTS 4096
#define KNN_K 20
#define NEG_SLOPE 0.2f
#define MTILE 4            // column tiles (of 16) per wave -> 64 columns

// ---------------------------------------------------------------------------
// Pack f32 weights (cout x cin) into f16 (coutp x cinp), zero padded.
// ---------------------------------------------------------------------------
__global__ void pack_w_kernel(const float* __restrict__ w, _Float16* __restrict__ out,
                              int coutp, int cout, int cin, int cinp)
{
    int e = blockIdx.x * blockDim.x + threadIdx.x;
    if (e >= coutp * cinp) return;
    int co = e / cinp, k = e % cinp;
    float v = (co < cout && k < cin) ? w[(size_t)co * cin + k] : 0.f;
    out[e] = (_Float16)v;
}

// ---------------------------------------------------------------------------
// KNN: per query point keep 20 nearest (squared distance), LDS-tiled candidates.
// ---------------------------------------------------------------------------
template <int C>
__global__ __launch_bounds__(256)
void knn_kernel(const float* __restrict__ pts, int Ctot, int coff,
                int* __restrict__ idx)
{
    __shared__ float sTile[C * 128];
    const int b  = blockIdx.y;
    const int nq = blockIdx.x * 256 + threadIdx.x;

    float q[C];
#pragma unroll
    for (int c = 0; c < C; ++c)
        q[c] = pts[((size_t)(b * Ctot + coff + c)) * NPTS + nq];

    float bestd[KNN_K];
    int   besti[KNN_K];
#pragma unroll
    for (int k = 0; k < KNN_K; ++k) { bestd[k] = 3.4e38f; besti[k] = 0; }

    for (int jb = 0; jb < NPTS; jb += 128) {
        __syncthreads();
        for (int t = threadIdx.x; t < C * 128; t += 256) {
            int c = t >> 7, j = t & 127;
            sTile[t] = pts[((size_t)(b * Ctot + coff + c)) * NPTS + jb + j];
        }
        __syncthreads();
        for (int j = 0; j < 128; ++j) {
            float d = 0.f;
#pragma unroll
            for (int c = 0; c < C; ++c) {
                float diff = q[c] - sTile[c * 128 + j];
                d += diff * diff;
            }
            if (d < bestd[KNN_K - 1]) {
                int p = KNN_K - 1;
                while (p > 0 && bestd[p - 1] > d) {
                    bestd[p] = bestd[p - 1]; besti[p] = besti[p - 1]; --p;
                }
                bestd[p] = d; besti[p] = jb + j;
            }
        }
    }
#pragma unroll
    for (int k = 0; k < KNN_K; ++k)
        idx[((size_t)(b * NPTS + nq)) * KNN_K + k] = besti[k];
}

// ---------------------------------------------------------------------------
// graph_feature: feat[m][0..C-1] = nbr - ctr, feat[m][C..2C-1] = ctr, padded.
// ---------------------------------------------------------------------------
__global__ void graph_feat_kernel(const float* __restrict__ pts, int C,
                                  const int* __restrict__ idx,
                                  _Float16* __restrict__ feat, int cinp)
{
    int m = blockIdx.x * blockDim.x + threadIdx.x;
    if (m >= 4 * NPTS * KNN_K) return;
    int n = (m / KNN_K) % NPTS, b = m / (NPTS * KNN_K);
    int j = idx[m];
    _Float16* row = feat + (size_t)m * cinp;
    for (int c = 0; c < C; ++c) {
        float ctr = pts[((size_t)(b * C + c)) * NPTS + n];
        float nbr = pts[((size_t)(b * C + c)) * NPTS + j];
        row[c]     = (_Float16)(nbr - ctr);
        row[C + c] = (_Float16)ctr;
    }
    for (int c = 2 * C; c < cinp; ++c) row[c] = (_Float16)0.f;
}

// ---------------------------------------------------------------------------
// WMMA GEMM: Y[cout][M] = QUAD ? (Wr X)*(Wg X + bg) + Wb (X*X)  :  Wr X
// One wave per block computes a 16 x 64 output block.
// Activation block X[m0..m0+63][0..cinp) is staged into LDS by the TDM
// (tensor_load_to_lds, D# per CDNA5 ISA 8.3/8.4); fragments load as b128.
// ---------------------------------------------------------------------------
template <bool QUAD, int SMODE>
__global__ __launch_bounds__(32)
void wmma_gemm_kernel(const _Float16* __restrict__ Wr, const _Float16* __restrict__ Wg,
                      const _Float16* __restrict__ Wb, const float* __restrict__ bg,
                      const _Float16* __restrict__ X, float* __restrict__ Yout,
                      int M, int cinp, int coutStore)
{
    extern __shared__ _Float16 sX[];                 // 64 x cinp f16 tile
    const int lane = threadIdx.x & 31;
    const int n = lane & 15, hi = lane >> 4;
    const int m0 = blockIdx.x * (16 * MTILE), r0 = blockIdx.y * 16;

    // ---- Tensor Data Mover: global X tile -> LDS -------------------------
    {
        uint64_t ga  = (uint64_t)(uintptr_t)(X + (size_t)m0 * cinp); // byte addr
        uint32_t lds = (uint32_t)(uintptr_t)sX;                      // LDS offset
        u32x4 g0;
        g0[0] = 1u;                                   // count=1 user descriptor
        g0[1] = lds;                                  // lds_addr
        g0[2] = (uint32_t)ga;                         // global_addr[31:0]
        g0[3] = (uint32_t)(ga >> 32) | (2u << 30);    // global_addr[56:32] | type=2
        u32x8 g1;
        g1[0] = 1u << 16;                             // data_size=1 (2 bytes)
        g1[1] = ((uint32_t)cinp & 0xffffu) << 16;     // tensor_dim0[15:0]
        g1[2] = (((uint32_t)cinp >> 16) & 0xffffu)    // tensor_dim0[31:16]
              | (((uint32_t)M & 0xffffu) << 16);      // tensor_dim1[15:0]
        g1[3] = (((uint32_t)M >> 16) & 0xffffu)       // tensor_dim1[31:16]
              | (((uint32_t)cinp & 0xffffu) << 16);   // tile_dim0 = cinp
        g1[4] = (uint32_t)(16 * MTILE);               // tile_dim1 = 64 rows
        g1[5] = (uint32_t)cinp;                       // tensor_dim0_stride[31:0]
        g1[6] = 0u;
        g1[7] = 0u;
        asm volatile("tensor_load_to_lds %0, %1" :: "s"(g0), "s"(g1) : "memory");
        __builtin_amdgcn_s_wait_tensorcnt(0);
    }

    v8f accR[MTILE], accG[MTILE], accB[MTILE];
#pragma unroll
    for (int t = 0; t < MTILE; ++t) { accR[t] = v8f{}; accG[t] = v8f{}; accB[t] = v8f{}; }

    for (int kc = 0; kc < cinp; kc += 32) {
        // A fragments (16x32 f16): halfs [kc+8hi .. +7] and [kc+16+8hi .. +7]
        // are contiguous -> two b128 loads per weight matrix.
        const size_t abase = (size_t)(r0 + n) * cinp + kc + 8 * hi;
        Frag ar, ag, ab;
        ar.q[0] = *(const uint4*)(Wr + abase);
        ar.q[1] = *(const uint4*)(Wr + abase + 16);
        if (QUAD) {
            ag.q[0] = *(const uint4*)(Wg + abase);
            ag.q[1] = *(const uint4*)(Wg + abase + 16);
            ab.q[0] = *(const uint4*)(Wb + abase);
            ab.q[1] = *(const uint4*)(Wb + abase + 16);
        }
        if (kc + 32 < cinp)
            __builtin_prefetch(Wr + abase + 32, 0, 0);

#pragma unroll
        for (int t = 0; t < MTILE; ++t) {
            // B fragment: halfs [kc+8hi .. +7] and [kc+16+8hi .. +7] at row (t*16+n)
            const _Float16* bp = sX + (size_t)(t * 16 + n) * cinp + kc + 8 * hi;
            Frag bx, bq;
            bx.q[0] = *(const uint4*)(bp);
            bx.q[1] = *(const uint4*)(bp + 16);
            if (QUAD) {
#pragma unroll
                for (int v = 0; v < 8; ++v) {
                    union { uint32_t u; _Float16 h[2]; } s; s.u = bx.u[v];
                    s.h[0] = s.h[0] * s.h[0];
                    s.h[1] = s.h[1] * s.h[1];
                    bq.u[v] = s.u;
                }
            }
            accR[t] = __builtin_amdgcn_wmma_f32_16x16x32_f16(false, ar.v, false, bx.v,
                                                             (short)0, accR[t], false, false);
            if (QUAD) {
                accG[t] = __builtin_amdgcn_wmma_f32_16x16x32_f16(false, ag.v, false, bx.v,
                                                                 (short)0, accG[t], false, false);
                accB[t] = __builtin_amdgcn_wmma_f32_16x16x32_f16(false, ab.v, false, bq.v,
                                                                 (short)0, accB[t], false, false);
            }
        }
    }

    float bgv[8];
#pragma unroll
    for (int j = 0; j < 8; ++j)
        bgv[j] = QUAD ? bg[r0 + j + 8 * hi] : 0.f;

#pragma unroll
    for (int t = 0; t < MTILE; ++t) {
#pragma unroll
        for (int j = 0; j < 8; ++j) {
            int row = r0 + j + 8 * hi;       // C/D layout: VGPR j -> M = j (+8 hi)
            int col = m0 + t * 16 + n;
            float y = accR[t][j];
            if (QUAD) y = y * (accG[t][j] + bgv[j]) + accB[t][j];
            if (SMODE == 0) {
                Yout[(size_t)row * M + col] = y;
            } else {
                int b = col >> 12, nn = col & 4095;
                if (row < coutStore)
                    Yout[((size_t)(b * coutStore + row) << 12) + nn] = y;
            }
        }
    }
}

// ---------------------------------------------------------------------------
// BN stats: one block per channel; stats[2c]=mean, stats[2c+1]=1/sqrt(var+eps)
// ---------------------------------------------------------------------------
__global__ __launch_bounds__(256)
void bn_stats_kernel(const float* __restrict__ Y, int M, float* __restrict__ stats)
{
    __shared__ float s1[256], s2[256];
    int c = blockIdx.x;
    const float* row = Y + (size_t)c * M;
    float s = 0.f, sq = 0.f;
    for (int i = threadIdx.x; i < M; i += 256) { float v = row[i]; s += v; sq += v * v; }
    s1[threadIdx.x] = s; s2[threadIdx.x] = sq;
    __syncthreads();
    for (int off = 128; off > 0; off >>= 1) {
        if (threadIdx.x < off) {
            s1[threadIdx.x] += s1[threadIdx.x + off];
            s2[threadIdx.x] += s2[threadIdx.x + off];
        }
        __syncthreads();
    }
    if (threadIdx.x == 0) {
        float mean = s1[0] / (float)M;
        float var  = s2[0] / (float)M - mean * mean;
        stats[2 * c]     = mean;
        stats[2 * c + 1] = rsqrtf(var + 1e-5f);
    }
}

__device__ __forceinline__ float bn_lrelu(float y, float mean, float is, float ga, float be)
{
    float v = (y - mean) * is * ga + be;
    return v >= 0.f ? v : NEG_SLOPE * v;
}

// BN+LeakyReLU -> f16 feature matrix [M][C] (input to next GEMM)
__global__ void bn_apply_feat_kernel(const float* __restrict__ Y, const float* __restrict__ stats,
                                     const float* __restrict__ gamma, const float* __restrict__ beta,
                                     _Float16* __restrict__ Xout, int M, int C)
{
    size_t e = (size_t)blockIdx.x * blockDim.x + threadIdx.x;
    if (e >= (size_t)M * C) return;
    int c = (int)(e / M);
    size_t m = e % M;
    float v = bn_lrelu(Y[e], stats[2 * c], stats[2 * c + 1], gamma[c], beta[c]);
    Xout[m * C + c] = (_Float16)v;
}

// BN+LeakyReLU then max over K neighbors -> x{1,2,3} [B][C][N] f32
__global__ void bn_maxk_kernel(const float* __restrict__ Y, const float* __restrict__ stats,
                               const float* __restrict__ gamma, const float* __restrict__ beta,
                               float* __restrict__ xout, int C)
{
    const int M = 4 * NPTS * KNN_K;
    int e = blockIdx.x * blockDim.x + threadIdx.x;
    if (e >= 4 * C * NPTS) return;
    int n = e % NPTS, c = (e / NPTS) % C, b = e / (NPTS * C);
    size_t mb = ((size_t)(b * NPTS + n)) * KNN_K;
    float mean = stats[2 * c], is = stats[2 * c + 1], ga = gamma[c], be = beta[c];
    float mx = -3.4e38f;
    for (int k = 0; k < KNN_K; ++k)
        mx = fmaxf(mx, bn_lrelu(Y[(size_t)c * M + mb + k], mean, is, ga, be));
    xout[e] = mx;   // e == (b*C+c)*N + n
}

// BN+LeakyReLU then max over N -> global feature g [B][1024]
__global__ __launch_bounds__(256)
void maxn_kernel(const float* __restrict__ Y, const float* __restrict__ stats,
                 const float* __restrict__ gamma, const float* __restrict__ beta,
                 float* __restrict__ g)
{
    __shared__ float sm[256];
    const int M = 4 * NPTS;
    int ch = blockIdx.x, b = blockIdx.y;
    float mean = stats[2 * ch], is = stats[2 * ch + 1], ga = gamma[ch], be = beta[ch];
    float mx = -3.4e38f;
    for (int n = threadIdx.x; n < NPTS; n += 256)
        mx = fmaxf(mx, bn_lrelu(Y[(size_t)ch * M + b * NPTS + n], mean, is, ga, be));
    sm[threadIdx.x] = mx; __syncthreads();
    for (int off = 128; off > 0; off >>= 1) {
        if (threadIdx.x < off) sm[threadIdx.x] = fmaxf(sm[threadIdx.x], sm[threadIdx.x + off]);
        __syncthreads();
    }
    if (threadIdx.x == 0) g[b * 1024 + ch] = sm[0];
}

// concat(x1,x2,x3) -> feat6 [M1][192] f16
__global__ void build_feat6_kernel(const float* __restrict__ x1, const float* __restrict__ x2,
                                   const float* __restrict__ x3, _Float16* __restrict__ feat)
{
    int m = blockIdx.x * blockDim.x + threadIdx.x;
    if (m >= 4 * NPTS) return;
    int b = m / NPTS, n = m % NPTS;
    _Float16* row = feat + (size_t)m * 192;
    for (int c = 0; c < 64; ++c) {
        row[c]       = (_Float16)x1[((size_t)(b * 64 + c)) * NPTS + n];
        row[64 + c]  = (_Float16)x2[((size_t)(b * 64 + c)) * NPTS + n];
        row[128 + c] = (_Float16)x3[((size_t)(b * 64 + c)) * NPTS + n];
    }
}

// concat(g_broadcast, x1,x2,x3) -> feat7 [M1][1216] f16
__global__ void build_feat7_kernel(const float* __restrict__ g, const float* __restrict__ x1,
                                   const float* __restrict__ x2, const float* __restrict__ x3,
                                   _Float16* __restrict__ feat)
{
    int m = blockIdx.x * blockDim.x + threadIdx.x;
    if (m >= 4 * NPTS) return;
    int b = m / NPTS, n = m % NPTS;
    _Float16* row = feat + (size_t)m * 1216;
    for (int c = 0; c < 1024; ++c) row[c] = (_Float16)g[b * 1024 + c];
    for (int c = 0; c < 64; ++c) {
        row[1024 + c]        = (_Float16)x1[((size_t)(b * 64 + c)) * NPTS + n];
        row[1024 + 64 + c]   = (_Float16)x2[((size_t)(b * 64 + c)) * NPTS + n];
        row[1024 + 128 + c]  = (_Float16)x3[((size_t)(b * 64 + c)) * NPTS + n];
    }
}

// ---------------------------------------------------------------------------
extern "C" void kernel_launch(void* const* d_in, const int* in_sizes, int n_in,
                              void* d_out, int out_size, void* d_ws, size_t ws_size,
                              hipStream_t stream)
{
    (void)in_sizes; (void)n_in; (void)out_size; (void)ws_size;
    const int B = 4;
    const size_t M2 = (size_t)B * NPTS * KNN_K;   // 327680 edges
    const size_t M1 = (size_t)B * NPTS;           // 16384 points

    const float* x = (const float*)d_in[0];
    auto W = [&](int i) { return (const float*)d_in[i]; };
    // c1:1-4 c2:5-8 c3:9-12 c4:13-16 c5:17-20 c6:21-24 (wr,wg,bg,wb)
    // w7:25 w8:26 w9:27 ; bn1g..bn8b : 28..43

    char* ws = (char*)d_ws;
    size_t off = 0;
    auto alloc = [&](size_t bytes) -> char* {
        char* p = ws + off;
        off = (off + bytes + 255) & ~(size_t)255;
        return p;
    };
    int*      idx   = (int*)     alloc(M2 * 4);
    _Float16* feat  = (_Float16*)alloc(M2 * 128 * 2);
    float*    Y2    = (float*)   alloc(64 * M2 * 4);
    float*    x1    = (float*)   alloc(M1 * 64 * 4);
    float*    x2    = (float*)   alloc(M1 * 64 * 4);
    float*    x3    = (float*)   alloc(M1 * 64 * 4);
    _Float16* feat6 = (_Float16*)alloc(M1 * 192 * 2);
    float*    Y6    = (float*)   alloc(1024 * M1 * 4);
    float*    gbuf  = (float*)   alloc(B * 1024 * 4);
    _Float16* feat7 = (_Float16*)alloc(M1 * 1216 * 2);
    float*    Y7    = (float*)   alloc(512 * M1 * 4);
    _Float16* feat8 = (_Float16*)alloc(M1 * 512 * 2);
    float*    Y8    = (float*)   alloc(256 * M1 * 4);
    _Float16* feat9 = (_Float16*)alloc(M1 * 256 * 2);
    _Float16* wp0   = (_Float16*)alloc((size_t)1024 * 1216 * 2);
    _Float16* wp1   = (_Float16*)alloc((size_t)1024 * 1216 * 2);
    _Float16* wp2   = (_Float16*)alloc((size_t)1024 * 1216 * 2);
    float*    stats = (float*)   alloc(2 * 1024 * 4);

    auto packw = [&](const float* w, _Float16* dst, int coutp, int cout, int cin, int cinp) {
        int tot = coutp * cinp;
        pack_w_kernel<<<(tot + 255) / 256, 256, 0, stream>>>(w, dst, coutp, cout, cin, cinp);
    };
    auto quad_gemm = [&](const float* bg, const _Float16* Xin, float* Yout,
                         int M, int cinp, int cout) {
        dim3 grid((unsigned)(M / (16 * MTILE)), (unsigned)(cout / 16));
        size_t sh = (size_t)(16 * MTILE) * cinp * sizeof(_Float16);
        wmma_gemm_kernel<true, 0><<<grid, 32, sh, stream>>>(wp0, wp1, wp2, bg, Xin, Yout,
                                                            M, cinp, 0);
    };
    auto plain_gemm = [&](const _Float16* Xin, float* Yout, int M, int cinp, int cout) {
        dim3 grid((unsigned)(M / (16 * MTILE)), (unsigned)(cout / 16));
        size_t sh = (size_t)(16 * MTILE) * cinp * sizeof(_Float16);
        wmma_gemm_kernel<false, 0><<<grid, 32, sh, stream>>>(wp0, wp0, wp0, nullptr, Xin, Yout,
                                                             M, cinp, 0);
    };
    auto final_gemm = [&](const _Float16* Xin, float* Yout, int M, int cinp, int cout,
                          int coutStore) {
        dim3 grid((unsigned)(M / (16 * MTILE)), (unsigned)(cout / 16));
        size_t sh = (size_t)(16 * MTILE) * cinp * sizeof(_Float16);
        wmma_gemm_kernel<false, 1><<<grid, 32, sh, stream>>>(wp0, wp0, wp0, nullptr, Xin, Yout,
                                                             M, cinp, coutStore);
    };
    int gM2 = (int)((M2 + 255) / 256);
    int gM1 = (int)((M1 + 255) / 256);

    // ================= stage 1: edge conv on xyz-knn =================
    knn_kernel<3><<<dim3(NPTS / 256, B), 256, 0, stream>>>(x, 9, 6, idx);
    graph_feat_kernel<<<gM2, 256, 0, stream>>>(x, 9, idx, feat, 32);

    packw(W(1), wp0, 64, 64, 18, 32); packw(W(2), wp1, 64, 64, 18, 32); packw(W(4), wp2, 64, 64, 18, 32);
    quad_gemm(W(3), feat, Y2, (int)M2, 32, 64);
    bn_stats_kernel<<<64, 256, 0, stream>>>(Y2, (int)M2, stats);
    bn_apply_feat_kernel<<<(unsigned)((64 * M2 + 255) / 256), 256, 0, stream>>>(
        Y2, stats, W(28), W(29), feat, (int)M2, 64);

    packw(W(5), wp0, 64, 64, 64, 64); packw(W(6), wp1, 64, 64, 64, 64); packw(W(8), wp2, 64, 64, 64, 64);
    quad_gemm(W(7), feat, Y2, (int)M2, 64, 64);
    bn_stats_kernel<<<64, 256, 0, stream>>>(Y2, (int)M2, stats);
    bn_maxk_kernel<<<(unsigned)((4 * 64 * NPTS + 255) / 256), 256, 0, stream>>>(
        Y2, stats, W(30), W(31), x1, 64);

    // ================= stage 2: edge conv on x1-knn ==================
    knn_kernel<64><<<dim3(NPTS / 256, B), 256, 0, stream>>>(x1, 64, 0, idx);
    graph_feat_kernel<<<gM2, 256, 0, stream>>>(x1, 64, idx, feat, 128);

    packw(W(9), wp0, 64, 64, 128, 128); packw(W(10), wp1, 64, 64, 128, 128); packw(W(12), wp2, 64, 64, 128, 128);
    quad_gemm(W(11), feat, Y2, (int)M2, 128, 64);
    bn_stats_kernel<<<64, 256, 0, stream>>>(Y2, (int)M2, stats);
    bn_apply_feat_kernel<<<(unsigned)((64 * M2 + 255) / 256), 256, 0, stream>>>(
        Y2, stats, W(32), W(33), feat, (int)M2, 64);

    packw(W(13), wp0, 64, 64, 64, 64); packw(W(14), wp1, 64, 64, 64, 64); packw(W(16), wp2, 64, 64, 64, 64);
    quad_gemm(W(15), feat, Y2, (int)M2, 64, 64);
    bn_stats_kernel<<<64, 256, 0, stream>>>(Y2, (int)M2, stats);
    bn_maxk_kernel<<<(unsigned)((4 * 64 * NPTS + 255) / 256), 256, 0, stream>>>(
        Y2, stats, W(34), W(35), x2, 64);

    // ================= stage 3: edge conv on x2-knn ==================
    knn_kernel<64><<<dim3(NPTS / 256, B), 256, 0, stream>>>(x2, 64, 0, idx);
    graph_feat_kernel<<<gM2, 256, 0, stream>>>(x2, 64, idx, feat, 128);

    packw(W(17), wp0, 64, 64, 128, 128); packw(W(18), wp1, 64, 64, 128, 128); packw(W(20), wp2, 64, 64, 128, 128);
    quad_gemm(W(19), feat, Y2, (int)M2, 128, 64);
    bn_stats_kernel<<<64, 256, 0, stream>>>(Y2, (int)M2, stats);
    bn_maxk_kernel<<<(unsigned)((4 * 64 * NPTS + 255) / 256), 256, 0, stream>>>(
        Y2, stats, W(36), W(37), x3, 64);

    // ================= stage 4: c6 + global max =====================
    build_feat6_kernel<<<gM1, 256, 0, stream>>>(x1, x2, x3, feat6);
    packw(W(21), wp0, 1024, 1024, 192, 192); packw(W(22), wp1, 1024, 1024, 192, 192); packw(W(24), wp2, 1024, 1024, 192, 192);
    quad_gemm(W(23), feat6, Y6, (int)M1, 192, 1024);
    bn_stats_kernel<<<1024, 256, 0, stream>>>(Y6, (int)M1, stats);
    maxn_kernel<<<dim3(1024, B), 256, 0, stream>>>(Y6, stats, W(38), W(39), gbuf);

    // ================= stage 5: head MLP ============================
    build_feat7_kernel<<<gM1, 256, 0, stream>>>(gbuf, x1, x2, x3, feat7);
    packw(W(25), wp0, 512, 512, 1216, 1216);
    plain_gemm(feat7, Y7, (int)M1, 1216, 512);
    bn_stats_kernel<<<512, 256, 0, stream>>>(Y7, (int)M1, stats);
    bn_apply_feat_kernel<<<(unsigned)((512 * M1 + 255) / 256), 256, 0, stream>>>(
        Y7, stats, W(40), W(41), feat8, (int)M1, 512);

    packw(W(26), wp0, 256, 256, 512, 512);
    plain_gemm(feat8, Y8, (int)M1, 512, 256);
    bn_stats_kernel<<<256, 256, 0, stream>>>(Y8, (int)M1, stats);
    bn_apply_feat_kernel<<<(unsigned)((256 * M1 + 255) / 256), 256, 0, stream>>>(
        Y8, stats, W(42), W(43), feat9, (int)M1, 256);

    packw(W(27), wp0, 16, 13, 256, 256);
    final_gemm(feat9, (float*)d_out, (int)M1, 256, 16, /*coutStore=*/13);
}
